// Attention3D_68633577390551
// MI455X (gfx1250) — matrix operand order repaired and checked by reading, not compiled, not run on hardware
//
#include <hip/hip_runtime.h>

// ---- types ----
typedef __attribute__((ext_vector_type(16))) __bf16          v16bf;
typedef __attribute__((ext_vector_type(8)))  float           v8f;
typedef __attribute__((ext_vector_type(8)))  unsigned short  v8us;
typedef __attribute__((ext_vector_type(4)))  unsigned int    v4u;
typedef __attribute__((ext_vector_type(4)))  int             v4i;
typedef __attribute__((ext_vector_type(8)))  int             v8i;

#define BB 8
#define CC 128
#define DD 16
#define KP 32        // k padded to 32 d-entries (upper 16 zero) -> maskless WMMA B
#define NN 4096      // 16*16*16 tokens
#define NT 256       // NN / 16 tiles

__device__ __forceinline__ unsigned short f2bf(float f) {
  unsigned int u = __float_as_uint(f);
  u += 0x7FFFu + ((u >> 16) & 1u);          // round-to-nearest-even
  return (unsigned short)(u >> 16);
}

union AB16 {
  v16bf v;
  v8us  h[2];
  unsigned short us[16];
};

// TDM: load a 2D bf16 tile [tile_d1 rows x tile_d0 elems] from a row-major
// tensor [tensor_d1 x tensor_d0] (row stride stride0 elems) into LDS.
// D# layout per CDNA5 ISA; 6-arg builtin on this toolchain.
__device__ __forceinline__ void tdm_load_2d(unsigned lds_off, const void* gaddr,
                                            unsigned tensor_d0, unsigned tensor_d1,
                                            unsigned tile_d0, unsigned tile_d1,
                                            unsigned stride0) {
  unsigned long long ga = (unsigned long long)gaddr;
  v4u g0 = { 1u,                                     // count=1, user descriptor
             lds_off,                                // lds_addr (bytes)
             (unsigned)(ga & 0xFFFFFFFFu),           // global_addr[31:0]
             (unsigned)((ga >> 32) & 0x01FFFFFFu) | (2u << 30) };  // addr[56:32], type=2
  v8i g1 = { (int)(1u << 16),                                   // data_size=1 -> 2B
             (int)((tensor_d0 & 0xFFFFu) << 16),                 // tensor_dim0 lo
             (int)(((tensor_d0 >> 16) & 0xFFFFu) |
                   ((tensor_d1 & 0xFFFFu) << 16)),               // dim0 hi | dim1 lo
             (int)(((tensor_d1 >> 16) & 0xFFFFu) |
                   ((tile_d0 & 0xFFFFu) << 16)),                 // dim1 hi | tile_dim0
             (int)(tile_d1 & 0xFFFFu),                           // tile_dim1 (tile_dim2=0)
             (int)stride0,                                       // tensor_dim0_stride lo
             0, 0 };
  v4i z4 = {0, 0, 0, 0};
  v8i z8 = {0, 0, 0, 0, 0, 0, 0, 0};
  __builtin_amdgcn_tensor_load_to_lds(g0, g1, z4, z4, z8, 0);
}

// =====================================================================
// Projections (1x1x1 convs = channel matmuls) -> bf16 operand buffers.
// q pre-scaled by log2(e) (softmax in exp2 domain); kT zero-padded to 32.
// =====================================================================
__global__ __launch_bounds__(256) void proj_qk_kernel(
    const float* __restrict__ x, const float* __restrict__ wq,
    const float* __restrict__ wk,
    unsigned short* __restrict__ qb, unsigned short* __restrict__ kb) {
  int idx = blockIdx.x * blockDim.x + threadIdx.x;   // over B*N
  if (idx >= BB * NN) return;
  int n = idx & (NN - 1);
  int b = idx >> 12;
  float aq[DD], ak[DD];
#pragma unroll
  for (int d = 0; d < DD; ++d) { aq[d] = 0.f; ak[d] = 0.f; }
  for (int c = 0; c < CC; ++c) {
    float xv = x[((size_t)(b * CC + c)) * NN + n];
#pragma unroll
    for (int d = 0; d < DD; ++d) {
      aq[d] = fmaf(wq[d * CC + c], xv, aq[d]);
      ak[d] = fmaf(wk[d * CC + c], xv, ak[d]);
    }
  }
  const float LOG2E = 1.44269504088896340736f;
  size_t oq = (size_t)idx * DD;
  size_t ok = (size_t)idx * KP;
#pragma unroll
  for (int d = 0; d < DD; ++d) {
    qb[oq + d] = f2bf(aq[d] * LOG2E);   // fold log2(e) into q
    kb[ok + d] = f2bf(ak[d]);
  }
#pragma unroll
  for (int d = DD; d < KP; ++d) kb[ok + d] = 0;   // zero pad: K>=16 lanes load 0
}

__global__ __launch_bounds__(256) void proj_v_kernel(
    const float* __restrict__ x, const float* __restrict__ wv,
    unsigned short* __restrict__ vb) {
  int idx = blockIdx.x * blockDim.x + threadIdx.x;   // over B*C*N
  if (idx >= BB * CC * NN) return;
  int n = idx & (NN - 1);
  int rest = idx >> 12;
  int c = rest & (CC - 1);
  int b = rest >> 7;
  float acc = 0.f;
  for (int c2 = 0; c2 < CC; ++c2)
    acc = fmaf(wv[c * CC + c2], x[((size_t)(b * CC + c2)) * NN + n], acc);
  vb[(size_t)idx] = f2bf(acc);   // layout (b*C + c)*N + n == idx
}

// =====================================================================
// Fused attention: one wave per (batch, 16-row query tile); 8 waves/block
// share one batch and cooperate on TDM-staged v/k tiles in pass 2.
// =====================================================================
__global__ __launch_bounds__(256) void attn_fused_kernel(
    const unsigned short* __restrict__ qb,
    const unsigned short* __restrict__ kb,
    const unsigned short* __restrict__ vb,
    const float* __restrict__ x,
    const float* __restrict__ gamma,
    float* __restrict__ out,      // [B, C, N]
    float* __restrict__ attn) {   // [B, N, N]
  __shared__ unsigned short attnTile[8 * 16 * 32];   // 1KB/wave: bf16 attn re-layout
  __shared__ unsigned short vTile[2][CC * 32];       // 2 x 8KB: v  [128 c][32 j]
  __shared__ unsigned short kTile[2][32 * KP];       // 2 x 2KB: kT [32 tok][32 d]

  const int wave  = threadIdx.x >> 5;
  const int lane  = threadIdx.x & 31;
  const int half  = lane >> 4;          // 0: lanes 0-15, 1: lanes 16-31
  const int l16   = lane & 15;
  const int g     = blockIdx.x * 8 + wave;      // 2048 waves total
  const int b     = g >> 8;                     // same b for all waves in block
  const int itile = g & 255;
  const int ibase = itile * 16;

  unsigned short* myLds = attnTile + wave * (16 * 32);

  // ---- A matrix for energy: q tile [16 x 32], K>=16 padded with zero ----
  AB16 aq;
  {
    const unsigned short* qrow =
        qb + ((size_t)(b * NN + ibase + l16)) * DD + half * 8;
    aq.h[0] = *(const v8us*)qrow;                 // e=0..7  -> K = 8*half + e
#pragma unroll
    for (int e = 8; e < 16; ++e) aq.us[e] = 0;    // e=8..15 -> K >= 16 (pad)
  }

  // ---------------- pass 1: online softmax stats (exp2 domain) ----------------
  float m[8], l[8];
#pragma unroll
  for (int r = 0; r < 8; ++r) { m[r] = -3.0e38f; l[r] = 0.f; }

  for (int jt = 0; jt < NT; ++jt) {
    AB16 bk;   // maskless: elements 8-15 of every lane read the zero-padded half of kb
    const unsigned short* krow =
        kb + ((size_t)(b * NN + jt * 16 + l16)) * KP + half * 8;
    bk.h[0] = *(const v8us*)krow;                 // e<8 -> K = 8*half + e, e>=8 -> K = 16 + 8*half + (e-8)
    bk.h[1] = *(const v8us*)(krow + 16);
    v8f z = {};
    v8f S = __builtin_amdgcn_wmma_f32_16x16x32_bf16(
        false, aq.v, false, bk.v, (short)0, z, false, false);
#pragma unroll
    for (int r = 0; r < 8; ++r) {
      float xv = S[r];
      float mn = fmaxf(m[r], xv);
      l[r] = l[r] * __builtin_amdgcn_exp2f(m[r] - mn) +
             __builtin_amdgcn_exp2f(xv - mn);
      m[r] = mn;
    }
  }
  // reduce (m,l) across the 16 lanes holding one row's columns
#pragma unroll
  for (int off = 1; off < 16; off <<= 1) {
#pragma unroll
    for (int r = 0; r < 8; ++r) {
      float mo = __shfl_xor(m[r], off, 32);
      float lo = __shfl_xor(l[r], off, 32);
      float M  = fmaxf(m[r], mo);
      l[r] = l[r] * __builtin_amdgcn_exp2f(m[r] - M) +
             lo * __builtin_amdgcn_exp2f(mo - M);
      m[r] = M;
    }
  }
  float invl[8];
#pragma unroll
  for (int r = 0; r < 8; ++r) invl[r] = __builtin_amdgcn_rcpf(l[r]);

  // ---------------- pass 2: TDM-staged, double-buffered ----------------
  v8f acc[8];
#pragma unroll
  for (int ct = 0; ct < 8; ++ct) acc[ct] = (v8f){};

  const unsigned short* vbase = vb + (size_t)b * CC * NN;   // [C][N]
  const unsigned short* kbase = kb + (size_t)b * NN * KP;   // [N][32]

  if (wave == 0) {   // prime buffer 0 with jp=0 tiles (uniform branch)
    tdm_load_2d((unsigned)(unsigned long long)&vTile[0][0], vbase, NN, CC, 32, CC, NN);
    tdm_load_2d((unsigned)(unsigned long long)&kTile[0][0], kbase, KP, NN, KP, 32, KP);
  }

  for (int jp = 0; jp < NT / 2; ++jp) {        // 32 j-columns per iteration
    const int cur = jp & 1;
    if (wave == 0) __builtin_amdgcn_s_wait_tensorcnt(0);   // tile[cur] resident
    __syncthreads();                                       // release to all waves
    if (wave == 0 && jp + 1 < NT / 2) {                    // prefetch next tile
      const size_t jn = (size_t)(jp + 1) * 32;
      tdm_load_2d((unsigned)(unsigned long long)&vTile[cur ^ 1][0],
                  vbase + jn, NN, CC, 32, CC, NN);
      tdm_load_2d((unsigned)(unsigned long long)&kTile[cur ^ 1][0],
                  kbase + jn * KP, KP, NN, KP, 32, KP);
    }

#pragma unroll
    for (int t = 0; t < 2; ++t) {
      const int jt = jp * 2 + t;
      AB16 bk;   // energy B matrix from LDS-staged kT tile (maskless)
      const unsigned short* krow = &kTile[cur][(t * 16 + l16) * KP + half * 8];
      bk.h[0] = *(const v8us*)krow;
      bk.h[1] = *(const v8us*)(krow + 16);
      v8f z = {};
      v8f S = __builtin_amdgcn_wmma_f32_16x16x32_bf16(
          false, aq.v, false, bk.v, (short)0, z, false, false);
#pragma unroll
      for (int r = 0; r < 8; ++r) {
        float a = __builtin_amdgcn_exp2f(S[r] - m[r]) * invl[r];
        const int ig = ibase + r + half * 8;          // D: M = r + 8*half
        const int jg = jt * 16 + l16;                 // D: N = lane%16
        attn[((size_t)b * NN + ig) * NN + jg] = a;    // single HBM write
        myLds[(r + half * 8) * 32 + t * 16 + l16] = f2bf(a);
      }
    }
    asm volatile("s_wait_dscnt 0" ::: "memory");      // wave-local re-layout fence

    // B matrix (attn^T role): lane = column i = l16, K = 32 j's
    AB16 ba;
    {
      const unsigned short* base = myLds + l16 * 32 + half * 8;
      ba.h[0] = *(const v8us*)base;
      ba.h[1] = *(const v8us*)(base + 16);
    }
    // out^T[c, i] += v[c, j] * attn[i, j]  -> A = LDS-staged v tile
#pragma unroll
    for (int ct = 0; ct < 8; ++ct) {
      AB16 av;
      const unsigned short* vrow = &vTile[cur][(ct * 16 + l16) * 32 + half * 8];
      av.h[0] = *(const v8us*)vrow;            // K = 8*half + (0..7)
      av.h[1] = *(const v8us*)(vrow + 16);     // K = 16 + 8*half + (0..7)
      acc[ct] = __builtin_amdgcn_wmma_f32_16x16x32_bf16(
          false, av.v, false, ba.v, (short)0, acc[ct], false, false);
    }
  }

  // ---- epilogue: out = gamma*out + x, coalesced along n ----
  const float gm = gamma[0];
#pragma unroll
  for (int ct = 0; ct < 8; ++ct) {
#pragma unroll
    for (int r = 0; r < 8; ++r) {
      const int c = ct * 16 + r + half * 8;    // D: M = c row
      const int n = ibase + l16;               // D: N = token index
      const size_t o = ((size_t)(b * CC + c)) * NN + n;
      out[o] = fmaf(gm, acc[ct][r], x[o]);
    }
  }
}

// =====================================================================
extern "C" void kernel_launch(void* const* d_in, const int* in_sizes, int n_in,
                              void* d_out, int out_size, void* d_ws, size_t ws_size,
                              hipStream_t stream) {
  const float* x     = (const float*)d_in[0];   // [8,128,16,16,16]
  const float* wq    = (const float*)d_in[1];   // [16,128]
  const float* wk    = (const float*)d_in[2];   // [16,128]
  const float* wv    = (const float*)d_in[3];   // [128,128]
  const float* gamma = (const float*)d_in[4];   // [1]

  float* out  = (float*)d_out;                         // [8,128,4096]
  float* attn = out + (size_t)BB * CC * NN;            // [8,4096,4096]

  unsigned short* qb = (unsigned short*)d_ws;          // [B,N,16] bf16
  unsigned short* kb = qb + (size_t)BB * NN * DD;      // [B,N,32] bf16 (padded)
  unsigned short* vb = kb + (size_t)BB * NN * KP;      // [B,C,N]  bf16
  // total ws use: 1MB + 2MB + 8MB = 11MB

  proj_qk_kernel<<<(BB * NN) / 256, 256, 0, stream>>>(x, wq, wk, qb, kb);
  proj_v_kernel<<<(BB * CC * NN) / 256, 256, 0, stream>>>(x, wv, vb);
  // 2048 waves = 8 waves/block * 256 blocks; one wave per (b, 16-row tile)
  attn_fused_kernel<<<256, 256, 0, stream>>>(qb, kb, vb, x, gamma, out, attn);
}
